// GroupedQueryAttention_42941083025421
// MI455X (gfx1250) — compile-verified, matrix operand order checked by reading, not run on hardware
//
#include <hip/hip_runtime.h>
#include <hip/hip_bf16.h>

#define N_EMBD 384
#define HQ     8
#define HKV    2
#define D_     48
#define G_     (HQ / HKV)   // 4
#define DP     64           // padded head dim
#define B_     8
#define T_     2048
#define BT     (B_ * T_)    // 16384

typedef __attribute__((ext_vector_type(16))) _Float16 v16h;
typedef __attribute__((ext_vector_type(8)))  _Float16 v8h;
typedef __attribute__((ext_vector_type(8)))  float    v8f;

// ---------------------------------------------------------------------------
// Fragment loader. For v_wmma_f32_16x16x32_f16 (wave32):
//   A (16x32, MxK): lane L holds row M=L&15; K-chunks (L>>4)*8 .. +7 in v0-3,
//                   and 16+(L>>4)*8 .. +7 in v4-7 (ISA 7.12.2).
//   B (32x16, KxN): same pattern with N in place of M, i.e. load from the
//                   row-major *transpose* [N,K] of B.
// Both reduce to: two contiguous 16-byte loads per lane from row-major [R,K].
// ---------------------------------------------------------------------------
static __device__ __forceinline__ v16h load_frag(const _Float16* p, int ld,
                                                 int r0, int k0) {
  int lane = threadIdx.x & 31;
  int r  = r0 + (lane & 15);
  int kb = k0 + ((lane >> 4) << 3);
  const _Float16* a = p + (size_t)r * ld + kb;
  v8h lo = *(const v8h*)(a);
  v8h hi = *(const v8h*)(a + 16);
  v16h f;
#pragma unroll
  for (int i = 0; i < 8; ++i) { f[i] = lo[i]; f[8 + i] = hi[i]; }
  return f;
}

static __device__ __forceinline__ v8f wmma16(v16h a, v16h b, v8f c) {
  return __builtin_amdgcn_wmma_f32_16x16x32_f16(false, a, false, b,
                                                (short)0, c, false, false);
}

// ---------------------------------------------------------------------------
// Utility kernels
// ---------------------------------------------------------------------------
__global__ void zero_u32_kernel(unsigned* p, long n) {
  long i = (long)blockIdx.x * blockDim.x + threadIdx.x;
  if (i < n) p[i] = 0u;
}

__global__ void cvt_f32_to_f16_kernel(_Float16* dst, const float* src, long n) {
  long i = (long)blockIdx.x * blockDim.x + threadIdx.x;
  if (i < n) dst[i] = (_Float16)src[i];
}

// dst[n*K + k] = (f16) src[k*N + n]   (weight [K,N] row-major -> [N,K] f16)
__global__ void transpose_w_kernel(_Float16* dst, const float* src, int K, int N) {
  long i = (long)blockIdx.x * blockDim.x + threadIdx.x;
  if (i >= (long)K * N) return;
  int n = (int)(i / K), k = (int)(i % K);
  dst[i] = (_Float16)src[(size_t)k * N + n];
}

// ---------------------------------------------------------------------------
// Q projection: q = (x @ wq) * 1/sqrt(D), scattered to [B,HKV,G,T,DP] f16
// One wave per 16x16 tile of the [BT, 384] output.
// ---------------------------------------------------------------------------
__global__ __launch_bounds__(256)
void proj_q_kernel(const _Float16* __restrict__ x16,
                   const _Float16* __restrict__ wqT,
                   _Float16* __restrict__ qbuf) {
  int wave = (int)((blockIdx.x * blockDim.x + threadIdx.x) >> 5);
  int lane = threadIdx.x & 31;
  const int NT = N_EMBD / 16;              // 24
  int m0 = (wave / NT) * 16;
  int n0 = (wave % NT) * 16;
  v8f c = {};
  for (int k = 0; k < N_EMBD; k += 32) {
    v16h a = load_frag(x16, N_EMBD, m0, k);
    v16h b = load_frag(wqT, N_EMBD, n0, k);
    c = wmma16(a, b, c);
  }
  const float scale = 0.14433756729740643f;   // 1/sqrt(48)
  int col = n0 + (lane & 15);
  int kvh = col / (G_ * D_);
  int g   = (col / D_) % G_;
  int d   = col % D_;
#pragma unroll
  for (int i = 0; i < 8; ++i) {
    int m = m0 + i + ((lane >> 4) << 3);
    int b_ = m / T_, t = m % T_;
    size_t off = ((((size_t)b_ * HKV + kvh) * G_ + g) * T_ + t) * DP + d;
    qbuf[off] = (_Float16)(c[i] * scale);
  }
}

// K projection -> [B,HKV,T,DP]; V projection -> V^T [B,HKV,DP,T]
__global__ __launch_bounds__(256)
void proj_kv_kernel(const _Float16* __restrict__ x16,
                    const _Float16* __restrict__ wT,
                    _Float16* __restrict__ obuf, int transposed) {
  int wave = (int)((blockIdx.x * blockDim.x + threadIdx.x) >> 5);
  int lane = threadIdx.x & 31;
  const int NT = (HKV * D_) / 16;          // 6
  int m0 = (wave / NT) * 16;
  int n0 = (wave % NT) * 16;
  v8f c = {};
  for (int k = 0; k < N_EMBD; k += 32) {
    v16h a = load_frag(x16, N_EMBD, m0, k);
    v16h b = load_frag(wT, N_EMBD, n0, k);
    c = wmma16(a, b, c);
  }
  int col = n0 + (lane & 15);
  int kvh = col / D_;
  int d   = col % D_;
#pragma unroll
  for (int i = 0; i < 8; ++i) {
    int m = m0 + i + ((lane >> 4) << 3);
    int b_ = m / T_, t = m % T_;
    size_t off;
    if (transposed)  // V^T: [B,HKV,DP,T]
      off = (((size_t)b_ * HKV + kvh) * DP + d) * T_ + t;
    else             // K:   [B,HKV,T,DP]
      off = (((size_t)b_ * HKV + kvh) * T_ + t) * DP + d;
    obuf[off] = (_Float16)c[i];
  }
}

// ---------------------------------------------------------------------------
// Flash attention: one wave per 16-row Q tile; online softmax; P staged in LDS.
// ---------------------------------------------------------------------------
__global__ __launch_bounds__(128)
void attn_kernel(const _Float16* __restrict__ qbuf,
                 const _Float16* __restrict__ kbuf,
                 const _Float16* __restrict__ vtbuf,
                 _Float16* __restrict__ aobuf) {
  __shared__ __align__(16) _Float16 pshared[4][16 * 32];
  int wlocal = threadIdx.x >> 5;
  int lane   = threadIdx.x & 31;
  int wave   = blockIdx.x * 4 + wlocal;
  const int QT = T_ / 16;                  // 128
  int qt = wave % QT;
  int h  = wave / QT;                      // (b*HKV + kvh)*G + g
  int g   = h % G_;
  int kvh = (h / G_) % HKV;
  int b   = h / (G_ * HKV);
  int q0  = qt * 16;

  const _Float16* qp = qbuf + ((((size_t)b * HKV + kvh) * G_ + g) * T_) * DP;
  const _Float16* kp = kbuf + (((size_t)b * HKV + kvh) * T_) * DP;
  const _Float16* vp = vtbuf + (((size_t)b * HKV + kvh) * DP) * T_;

  v16h qf0 = load_frag(qp, DP, q0, 0);
  v16h qf1 = load_frag(qp, DP, q0, 32);

  v8f o0 = {}, o1 = {}, o2 = {}, o3 = {};
  float m_[8], l_[8];
#pragma unroll
  for (int i = 0; i < 8; ++i) { m_[i] = -1e30f; l_[i] = 0.f; }

  int col   = lane & 15;
  int rbase = (lane >> 4) << 3;
  _Float16* pb = &pshared[wlocal][0];

  int nchunks = (q0 + 47) >> 5;            // covers tk <= q0+15
  for (int kc = 0; kc < nchunks; ++kc) {
    int kb = kc << 5;
    v8f s0 = {}, s1 = {};
    {
      v16h b0 = load_frag(kp, DP, kb, 0);
      v16h b1 = load_frag(kp, DP, kb, 32);
      s0 = wmma16(qf0, b0, s0);
      s0 = wmma16(qf1, b1, s0);
    }
    {
      v16h b0 = load_frag(kp, DP, kb + 16, 0);
      v16h b1 = load_frag(kp, DP, kb + 16, 32);
      s1 = wmma16(qf0, b0, s1);
      s1 = wmma16(qf1, b1, s1);
    }
    // causal mask + online softmax
#pragma unroll
    for (int i = 0; i < 8; ++i) {
      int tq = q0 + i + rbase;
      if (kb + col > tq)      s0[i] = -1e30f;
      if (kb + 16 + col > tq) s1[i] = -1e30f;
      float r = fmaxf(s0[i], s1[i]);
      for (int off = 1; off < 16; off <<= 1)
        r = fmaxf(r, __shfl_xor(r, off, 32));
      float mn   = fmaxf(m_[i], r);
      float corr = __expf(m_[i] - mn);
      float p0   = __expf(s0[i] - mn);
      float p1   = __expf(s1[i] - mn);
      float rs   = p0 + p1;
      for (int off = 1; off < 16; off <<= 1)
        rs += __shfl_xor(rs, off, 32);
      l_[i] = l_[i] * corr + rs;
      m_[i] = mn;
      o0[i] *= corr; o1[i] *= corr; o2[i] *= corr; o3[i] *= corr;
      int row = i + rbase;
      pb[row * 32 + col]      = (_Float16)p0;
      pb[row * 32 + 16 + col] = (_Float16)p1;
    }
    // P [16x32] from LDS as A-fragment; V^T rows = d, cols = t
    v16h pa  = load_frag(pb, 32, 0, 0);
    v16h vb0 = load_frag(vp, T_, 0,  kb);
    v16h vb1 = load_frag(vp, T_, 16, kb);
    v16h vb2 = load_frag(vp, T_, 32, kb);
    v16h vb3 = load_frag(vp, T_, 48, kb);
    o0 = wmma16(pa, vb0, o0);
    o1 = wmma16(pa, vb1, o1);
    o2 = wmma16(pa, vb2, o2);
    o3 = wmma16(pa, vb3, o3);
  }

  // epilogue: O /= l, write [BT, HQ*D] f16 (only d < 48 valid)
  int hq = kvh * G_ + g;
#pragma unroll
  for (int i = 0; i < 8; ++i) {
    float inv = 1.0f / l_[i];
    int t = q0 + i + rbase;
    size_t rowoff = ((size_t)b * T_ + t) * (HQ * D_) + (size_t)hq * D_;
    float vals[4] = { o0[i] * inv, o1[i] * inv, o2[i] * inv, o3[i] * inv };
#pragma unroll
    for (int nt = 0; nt < 4; ++nt) {
      int d = nt * 16 + col;
      if (d < D_) aobuf[rowoff + d] = (_Float16)vals[nt];
    }
  }
}

// ---------------------------------------------------------------------------
// Output projection: out = aobuf @ wo + bo  (f32 result)
// ---------------------------------------------------------------------------
__global__ __launch_bounds__(256)
void out_proj_kernel(const _Float16* __restrict__ aobuf,
                     const _Float16* __restrict__ woT,
                     const float* __restrict__ bo,
                     float* __restrict__ out) {
  int wave = (int)((blockIdx.x * blockDim.x + threadIdx.x) >> 5);
  int lane = threadIdx.x & 31;
  const int NT = N_EMBD / 16;
  int m0 = (wave / NT) * 16;
  int n0 = (wave % NT) * 16;
  v8f c = {};
  for (int k = 0; k < N_EMBD; k += 32) {
    v16h a = load_frag(aobuf, N_EMBD, m0, k);
    v16h b = load_frag(woT, N_EMBD, n0, k);
    c = wmma16(a, b, c);
  }
  int n = n0 + (lane & 15);
  float bias = bo[n];
#pragma unroll
  for (int i = 0; i < 8; ++i) {
    int m = m0 + i + ((lane >> 4) << 3);
    out[(size_t)m * N_EMBD + n] = c[i] + bias;
  }
}

// ---------------------------------------------------------------------------
// Host launcher
// ---------------------------------------------------------------------------
extern "C" void kernel_launch(void* const* d_in, const int* in_sizes, int n_in,
                              void* d_out, int out_size, void* d_ws, size_t ws_size,
                              hipStream_t stream) {
  const float* x  = (const float*)d_in[0];
  const float* wq = (const float*)d_in[1];
  const float* wk = (const float*)d_in[2];
  const float* wv = (const float*)d_in[3];
  const float* wo = (const float*)d_in[4];
  const float* bo = (const float*)d_in[5];
  float* out = (float*)d_out;

  // workspace layout (f16 buffers), 256B aligned
  char* ws = (char*)d_ws;
  size_t off = 0;
  auto alloc = [&](size_t bytes) {
    char* p = ws + off;
    off += (bytes + 255) & ~(size_t)255;
    return p;
  };
  _Float16* x16   = (_Float16*)alloc((size_t)BT * N_EMBD * 2);      // 12.6 MB
  _Float16* wqT   = (_Float16*)alloc((size_t)N_EMBD * N_EMBD * 2);  // [384,384]
  _Float16* wkT   = (_Float16*)alloc((size_t)HKV * D_ * N_EMBD * 2);// [96,384]
  _Float16* wvT   = (_Float16*)alloc((size_t)HKV * D_ * N_EMBD * 2);
  _Float16* woT   = (_Float16*)alloc((size_t)N_EMBD * N_EMBD * 2);
  _Float16* qbuf  = (_Float16*)alloc((size_t)B_ * HKV * G_ * T_ * DP * 2); // 16.8 MB
  _Float16* kbuf  = (_Float16*)alloc((size_t)B_ * HKV * T_ * DP * 2);      // 4.2 MB
  _Float16* vtbuf = (_Float16*)alloc((size_t)B_ * HKV * DP * T_ * 2);      // 4.2 MB
  _Float16* aobuf = (_Float16*)alloc((size_t)BT * HQ * D_ * 2);            // 12.6 MB
  (void)ws_size; (void)in_sizes; (void)n_in; (void)out_size;

  // 1) zero q/k/vt (covers the DP padding lanes, deterministic each call)
  {
    long ndw = ((long)B_ * HKV * G_ * T_ * DP +
                (long)B_ * HKV * T_ * DP +
                (long)B_ * HKV * DP * T_) * 2 / 4;
    int blk = 256;
    zero_u32_kernel<<<dim3((unsigned)((ndw + blk - 1) / blk)), dim3(blk), 0, stream>>>(
        (unsigned*)qbuf, ndw);
  }
  // 2) convert x -> f16
  {
    long n = (long)BT * N_EMBD;
    cvt_f32_to_f16_kernel<<<dim3((unsigned)((n + 255) / 256)), dim3(256), 0, stream>>>(
        x16, x, n);
  }
  // 3) transpose+convert weights to [N,K] f16
  {
    long n = (long)N_EMBD * N_EMBD;
    transpose_w_kernel<<<dim3((unsigned)((n + 255) / 256)), dim3(256), 0, stream>>>(
        wqT, wq, N_EMBD, N_EMBD);
    transpose_w_kernel<<<dim3((unsigned)((n + 255) / 256)), dim3(256), 0, stream>>>(
        woT, wo, N_EMBD, N_EMBD);
    long nk = (long)N_EMBD * HKV * D_;
    transpose_w_kernel<<<dim3((unsigned)((nk + 255) / 256)), dim3(256), 0, stream>>>(
        wkT, wk, N_EMBD, HKV * D_);
    transpose_w_kernel<<<dim3((unsigned)((nk + 255) / 256)), dim3(256), 0, stream>>>(
        wvT, wv, N_EMBD, HKV * D_);
  }
  // 4) projections
  {
    int waves_q = (BT / 16) * (N_EMBD / 16);        // 24576
    proj_q_kernel<<<dim3(waves_q * 32 / 256), dim3(256), 0, stream>>>(x16, wqT, qbuf);
    int waves_kv = (BT / 16) * (HKV * D_ / 16);     // 6144
    proj_kv_kernel<<<dim3(waves_kv * 32 / 256), dim3(256), 0, stream>>>(x16, wkT, kbuf, 0);
    proj_kv_kernel<<<dim3(waves_kv * 32 / 256), dim3(256), 0, stream>>>(x16, wvT, vtbuf, 1);
  }
  // 5) attention: one wave per 16-row q tile, 4 waves/block
  {
    int waves = B_ * HKV * G_ * (T_ / 16);          // 8192
    attn_kernel<<<dim3(waves / 4), dim3(128), 0, stream>>>(qbuf, kbuf, vtbuf, aobuf);
  }
  // 6) output projection + bias
  {
    int waves = (BT / 16) * (N_EMBD / 16);          // 24576
    out_proj_kernel<<<dim3(waves * 32 / 256), dim3(256), 0, stream>>>(aobuf, woT, bo, out);
  }
}
